// MultiHeadAttention_89773406421150
// MI455X (gfx1250) — matrix operand order repaired and checked by reading, not compiled, not run on hardware
//
#include <hip/hip_runtime.h>
#include <hip/hip_bf16.h>

typedef __attribute__((ext_vector_type(16))) _Float16 v16h;
typedef __attribute__((ext_vector_type(8)))  _Float16 v8h;
typedef __attribute__((ext_vector_type(4)))  _Float16 v4h;
typedef __attribute__((ext_vector_type(8)))  float    v8f;
typedef __attribute__((ext_vector_type(4)))  float    v4f;
typedef __attribute__((ext_vector_type(4)))  int      v4i;

#if defined(__has_builtin)
#if __has_builtin(__builtin_amdgcn_global_load_async_to_lds_b128) && \
    __has_builtin(__builtin_amdgcn_s_wait_asynccnt)
#define HAVE_ASYNC_LDS 1
#endif
#endif

#ifdef HAVE_ASYNC_LDS
// Builtin expects v4i pointers in the global (AS1) and LDS (AS3) address
// spaces. Generic->AS casts via integers: generic global ptrs are numerically
// AS1; generic LDS ptrs carry the LDS offset in their low 32 bits.
typedef __attribute__((address_space(1))) v4i gv4i;
typedef __attribute__((address_space(3))) v4i lv4i;
__device__ __forceinline__ gv4i* to_global(const void* p) {
    return (gv4i*)(unsigned long long)(uintptr_t)p;
}
__device__ __forceinline__ lv4i* to_lds(const void* p) {
    return (lv4i*)(unsigned int)(uintptr_t)p;
}
#endif

// ---------------------------------------------------------------------------
// WMMA f16 16x16x32 fragment loaders (wave32, CDNA5 ISA layouts).
// A 16x32: lane=M%16, K halves {kb..kb+7, kb+16..kb+23}, kb=(lane>=16)*8.
__device__ __forceinline__ v16h load_fragA(const _Float16* p) {
    union { v16h v; v8h h[2]; } u;
    u.h[0] = *(const v8h*)(p);
    u.h[1] = *(const v8h*)(p + 16);
    return u.v;
}
// B 32x16: lane=N%16, K halves {kb..kb+7, kb+16..kb+23}, kb=(lane>=16)*8 (LDS tile is [N][K]).
__device__ __forceinline__ v16h load_fragB(const _Float16* p) {
    union { v16h v; v8h h[2]; } u;
    u.h[0] = *(const v8h*)(p);
    u.h[1] = *(const v8h*)(p + 16);
    return u.v;
}
__device__ __forceinline__ v8f wmma16(v16h a, v16h b, v8f c) {
    return __builtin_amdgcn_wmma_f32_16x16x32_f16(false, a, false, b,
                                                  (short)0, c, false, false);
}
__device__ __forceinline__ v4h cvt4(v4f a) {
    v4h r;
    r.x = (_Float16)a.x; r.y = (_Float16)a.y;
    r.z = (_Float16)a.z; r.w = (_Float16)a.w;
    return r;
}

// ---------------------------------------------------------------------------
// Register-staged tile movers (128x32 A tile, 32x128 B tile, BK = 32)
__device__ __forceinline__ void a32_load(const float* __restrict__ A, size_t ld,
                                         int bm, int kk, int tid, v4f ra[4]) {
    #pragma unroll
    for (int t = 0; t < 4; t++) {
        int idx = tid + t * 256;                  // 1024 x v4f
        int m = idx >> 3, kq = (idx & 7) << 2;
        ra[t] = *(const v4f*)&A[(size_t)(bm + m) * ld + kk + kq];
    }
}
__device__ __forceinline__ void a32_load_nt(const float* __restrict__ A, size_t ld,
                                            int bm, int kk, int tid, v4f ra[4]) {
    #pragma unroll
    for (int t = 0; t < 4; t++) {
        int idx = tid + t * 256;
        int m = idx >> 3, kq = (idx & 7) << 2;
        ra[t] = __builtin_nontemporal_load(
            (const v4f*)&A[(size_t)(bm + m) * ld + kk + kq]);
    }
}
__device__ __forceinline__ void a32_stage(_Float16* As, int tid, const v4f ra[4]) {
    #pragma unroll
    for (int t = 0; t < 4; t++) {
        int idx = tid + t * 256;
        int m = idx >> 3, kq = (idx & 7) << 2;
        *(v4h*)&As[m * 32 + kq] = cvt4(ra[t]);    // 8B aligned
    }
}
__device__ __forceinline__ void w_load(const float* __restrict__ W,
                                       int bn, int kk, int tid, v4f rb[4]) {
    #pragma unroll
    for (int t = 0; t < 4; t++) {
        int idx = tid + t * 256;                  // 32 k-rows x 128 n
        int kq = idx >> 5, n = (idx & 31) << 2;
        rb[t] = *(const v4f*)&W[(size_t)(kk + kq) * 1024 + bn + n];
    }
}
__device__ __forceinline__ void w_stage(_Float16* Bs, int tid, const v4f rb[4]) {
    #pragma unroll
    for (int t = 0; t < 4; t++) {
        int idx = tid + t * 256;
        int kq = idx >> 5, n = (idx & 31) << 2;
        Bs[(n + 0) * 32 + kq] = (_Float16)rb[t].x;   // transpose to [n][k]
        Bs[(n + 1) * 32 + kq] = (_Float16)rb[t].y;
        Bs[(n + 2) * 32 + kq] = (_Float16)rb[t].z;
        Bs[(n + 3) * 32 + kq] = (_Float16)rb[t].w;
    }
}

// One BK=32 slab of 2x4 16x16 WMMA tiles per wave.
__device__ __forceinline__ void mma_2x4(const _Float16* As, const _Float16* Bs,
                                        int wm, int wn, int r, int ka, int kbv,
                                        v8f (&acc)[2][4]) {
    v16h af[2], bf[4];
    #pragma unroll
    for (int i = 0; i < 2; i++) af[i] = load_fragA(&As[(wm + i * 16 + r) * 32 + ka]);
    #pragma unroll
    for (int j = 0; j < 4; j++) bf[j] = load_fragB(&Bs[(wn + j * 16 + r) * 32 + kbv]);
    #pragma unroll
    for (int i = 0; i < 2; i++)
        #pragma unroll
        for (int j = 0; j < 4; j++)
            acc[i][j] = wmma16(af[i], bf[j], acc[i][j]);
}

// ---------------------------------------------------------------------------
// Projection GEMM: O[B*H,S,64](f16) = (X[4096,1024] @ W + b) * scale
__global__ __launch_bounds__(256) void proj_kernel(
    const float* __restrict__ X, const float* __restrict__ W,
    const float* __restrict__ bias, float scale, _Float16* __restrict__ O)
{
    __shared__ alignas(16) _Float16 As[128 * 32];
    __shared__ alignas(16) _Float16 Bs[128 * 32];
    const int tid  = threadIdx.x;
    const int lane = tid & 31, wave = tid >> 5;
    const int bm = blockIdx.y * 128, bn = blockIdx.x * 128;
    const int wm = (wave >> 1) * 32, wn = (wave & 1) * 64;
    const int r  = lane & 15;
    const int ka = (lane >> 4) * 8, kbv = (lane >> 4) * 8;

    v8f acc[2][4];
    v8f vz = {};
    #pragma unroll
    for (int i = 0; i < 2; i++)
        #pragma unroll
        for (int j = 0; j < 4; j++) acc[i][j] = vz;

    v4f ra0[4], rb0[4], ra1[4], rb1[4];
    a32_load(X, 1024, bm, 0, tid, ra0);
    w_load(W, bn, 0, tid, rb0);

    for (int kk = 0; kk < 1024; kk += 64) {
        a32_stage(As, tid, ra0);
        w_stage(Bs, tid, rb0);
        __syncthreads();
        a32_load(X, 1024, bm, kk + 32, tid, ra1);   // kk+32 <= 992 always valid
        w_load(W, bn, kk + 32, tid, rb1);
        mma_2x4(As, Bs, wm, wn, r, ka, kbv, acc);
        __syncthreads();
        a32_stage(As, tid, ra1);
        w_stage(Bs, tid, rb1);
        __syncthreads();
        if (kk + 64 < 1024) {
            a32_load(X, 1024, bm, kk + 64, tid, ra0);
            w_load(W, bn, kk + 64, tid, rb0);
        }
        mma_2x4(As, Bs, wm, wn, r, ka, kbv, acc);
        __syncthreads();
    }

    const int rbase = (lane >> 4) * 8;
    #pragma unroll
    for (int i = 0; i < 2; i++)
        #pragma unroll
        for (int j = 0; j < 4; j++)
            #pragma unroll
            for (int e = 0; e < 8; e++) {
                int m = bm + wm + i * 16 + rbase + e;
                int n = bn + wn + j * 16 + r;
                float val = (acc[i][j][e] + bias[n]) * scale;
                int b = m >> 11, s = m & 2047;
                int h = n >> 6,  d = n & 63;
                O[(((size_t)(b * 16 + h)) * 2048 + s) * 64 + d] = (_Float16)val;
            }
}

// ---------------------------------------------------------------------------
// Scores: attn[bh] = Qs[bh] @ K[bh]^T  (Q pre-scaled by 1/8). Tile 128x128, K=64.
__global__ __launch_bounds__(256) void scores_kernel(
    const _Float16* __restrict__ Qh, const _Float16* __restrict__ Kh,
    float* __restrict__ attn)
{
    __shared__ alignas(16) _Float16 Qs[128 * 64];
    __shared__ alignas(16) _Float16 Ks[128 * 64];
    const int tid = threadIdx.x, lane = tid & 31, wave = tid >> 5;
    const int bh = blockIdx.z;
    const int bm = blockIdx.y * 128, bn = blockIdx.x * 128;
    const _Float16* Qb = Qh + (size_t)bh * 2048 * 64;
    const _Float16* Kb = Kh + (size_t)bh * 2048 * 64;
    float* Ab = attn + (size_t)bh * 2048 * 2048;

#ifdef HAVE_ASYNC_LDS
    {   // ASYNCcnt-tracked direct global->LDS copy, no VGPR round trip
        const _Float16* gq = Qb + (size_t)bm * 64;
        const _Float16* gk = Kb + (size_t)bn * 64;
        #pragma unroll
        for (int t = 0; t < 4; t++) {
            int hidx = (tid + t * 256) * 8;       // 16B chunks of halves
            __builtin_amdgcn_global_load_async_to_lds_b128(
                to_global(gq + hidx), to_lds(Qs + hidx), 0, 0);
            __builtin_amdgcn_global_load_async_to_lds_b128(
                to_global(gk + hidx), to_lds(Ks + hidx), 0, 0);
        }
        __builtin_amdgcn_s_wait_asynccnt(0);
    }
#else
    {   // contiguous 16 KB tiles: bulk 16B copies
        const uint4* sq = (const uint4*)(Qb + (size_t)bm * 64);
        const uint4* sk = (const uint4*)(Kb + (size_t)bn * 64);
        uint4* dq = (uint4*)Qs;
        uint4* dk = (uint4*)Ks;
        #pragma unroll
        for (int t = 0; t < 4; t++) dq[tid + t * 256] = sq[tid + t * 256];
        #pragma unroll
        for (int t = 0; t < 4; t++) dk[tid + t * 256] = sk[tid + t * 256];
    }
#endif
    __syncthreads();

    const int wm = (wave >> 1) * 32, wn = (wave & 1) * 64;
    const int r = lane & 15;
    const int ka = (lane >> 4) * 8, kbv = (lane >> 4) * 8;
    v8f acc[2][4];
    v8f vz = {};
    #pragma unroll
    for (int i = 0; i < 2; i++)
        #pragma unroll
        for (int j = 0; j < 4; j++) acc[i][j] = vz;

    #pragma unroll
    for (int ks = 0; ks < 2; ks++) {
        v16h af[2], bf[4];
        #pragma unroll
        for (int i = 0; i < 2; i++)
            af[i] = load_fragA(&Qs[(wm + i * 16 + r) * 64 + ks * 32 + ka]);
        #pragma unroll
        for (int j = 0; j < 4; j++)
            bf[j] = load_fragB(&Ks[(wn + j * 16 + r) * 64 + ks * 32 + kbv]);
        #pragma unroll
        for (int i = 0; i < 2; i++)
            #pragma unroll
            for (int j = 0; j < 4; j++)
                acc[i][j] = wmma16(af[i], bf[j], acc[i][j]);
    }

    const int rbase = (lane >> 4) * 8;
    #pragma unroll
    for (int i = 0; i < 2; i++)
        #pragma unroll
        for (int j = 0; j < 4; j++)
            #pragma unroll
            for (int e = 0; e < 8; e++) {
                int m = bm + wm + i * 16 + rbase + e;
                int n = bn + wn + j * 16 + r;
                __builtin_nontemporal_store(acc[i][j][e],
                                            &Ab[(size_t)m * 2048 + n]);
            }
}

// ---------------------------------------------------------------------------
// Row softmax over 2048 floats: wave32 shuffle reductions, 2 barriers total.
__global__ __launch_bounds__(256) void softmax_kernel(float* __restrict__ attn)
{
    __shared__ float redmax[8];
    __shared__ float redsum[8];
    const int tid = threadIdx.x, lane = tid & 31, wave = tid >> 5;
    float* row = attn + (size_t)blockIdx.x * 2048;

    v4f vals[2];
    float m = -3.402823e38f;
    #pragma unroll
    for (int t = 0; t < 2; t++) {
        vals[t] = __builtin_nontemporal_load((const v4f*)&row[(tid + t * 256) * 4]);
        #pragma unroll
        for (int i = 0; i < 4; i++) m = fmaxf(m, vals[t][i]);
    }
    #pragma unroll
    for (int off = 16; off > 0; off >>= 1)
        m = fmaxf(m, __shfl_xor(m, off, 32));
    if (lane == 0) redmax[wave] = m;
    __syncthreads();
    #pragma unroll
    for (int w = 0; w < 8; w++) m = fmaxf(m, redmax[w]);

    float s = 0.0f;
    #pragma unroll
    for (int t = 0; t < 2; t++)
        #pragma unroll
        for (int i = 0; i < 4; i++) {
            vals[t][i] = __expf(vals[t][i] - m);
            s += vals[t][i];
        }
    #pragma unroll
    for (int off = 16; off > 0; off >>= 1)
        s += __shfl_xor(s, off, 32);
    if (lane == 0) redsum[wave] = s;
    __syncthreads();
    s = 0.0f;
    #pragma unroll
    for (int w = 0; w < 8; w++) s += redsum[w];

    float inv = 1.0f / s;
    #pragma unroll
    for (int t = 0; t < 2; t++) {
        v4f o = vals[t] * inv;
        __builtin_nontemporal_store(o, (v4f*)&row[(tid + t * 256) * 4]);
    }
}

// ---------------------------------------------------------------------------
// P@V per (b,h): M=2048 (128/block), N=64, K=2048 (BK=32), double-buffered.
__device__ __forceinline__ void avV_load(const _Float16* __restrict__ Vb,
                                         int kk, int tid, v8h& rv) {
    int k = tid >> 3, n = (tid & 7) << 3;         // 32x64 tile = 256 x v8h
    rv = *(const v8h*)&Vb[(size_t)(kk + k) * 64 + n];
}
__device__ __forceinline__ void avV_stage(_Float16* Vs, int tid, v8h rv) {
    int k = tid >> 3, n = (tid & 7) << 3;
    #pragma unroll
    for (int i = 0; i < 8; i++) Vs[(n + i) * 32 + k] = rv[i];   // [n][k]
}
__device__ __forceinline__ void mma_1x4(const _Float16* As, const _Float16* Vs,
                                        int wrow, int r, int ka, int kbv,
                                        v8f (&acc)[4]) {
    v16h af = load_fragA(&As[(wrow + r) * 32 + ka]);
    #pragma unroll
    for (int j = 0; j < 4; j++) {
        v16h bf = load_fragB(&Vs[(j * 16 + r) * 32 + kbv]);
        acc[j] = wmma16(af, bf, acc[j]);
    }
}

__global__ __launch_bounds__(256) void av_kernel(
    const float* __restrict__ attn, const _Float16* __restrict__ Vh,
    _Float16* __restrict__ ctx)
{
    __shared__ alignas(16) _Float16 As[128 * 32];   // P tile [m][k]
    __shared__ alignas(16) _Float16 Vs[64 * 32];    // V tile [n][k]
    const int tid = threadIdx.x, lane = tid & 31, wave = tid >> 5;
    const int bh = blockIdx.z;
    const int bm = blockIdx.x * 128;
    const float* Ab = attn + (size_t)bh * 2048 * 2048;
    const _Float16* Vb = Vh + (size_t)bh * 2048 * 64;
    const int r = lane & 15;
    const int ka = (lane >> 4) * 8, kbv = (lane >> 4) * 8;
    const int wrow = wave * 16;

    v8f acc[4];
    v8f vz = {};
    #pragma unroll
    for (int j = 0; j < 4; j++) acc[j] = vz;

    v4f rp0[4], rp1[4];
    v8h rv0, rv1;
    a32_load_nt(Ab, 2048, bm, 0, tid, rp0);
    avV_load(Vb, 0, tid, rv0);

    for (int kk = 0; kk < 2048; kk += 64) {
        a32_stage(As, tid, rp0);
        avV_stage(Vs, tid, rv0);
        __syncthreads();
        a32_load_nt(Ab, 2048, bm, kk + 32, tid, rp1);   // kk+32 <= 2016
        avV_load(Vb, kk + 32, tid, rv1);
        mma_1x4(As, Vs, wrow, r, ka, kbv, acc);
        __syncthreads();
        a32_stage(As, tid, rp1);
        avV_stage(Vs, tid, rv1);
        __syncthreads();
        if (kk + 64 < 2048) {
            a32_load_nt(Ab, 2048, bm, kk + 64, tid, rp0);
            avV_load(Vb, kk + 64, tid, rv0);
        }
        mma_1x4(As, Vs, wrow, r, ka, kbv, acc);
        __syncthreads();
    }

    const int rbase = (lane >> 4) * 8;
    const int b = bh >> 4, h = bh & 15;
    #pragma unroll
    for (int j = 0; j < 4; j++)
        #pragma unroll
        for (int e = 0; e < 8; e++) {
            int m = bm + wrow + rbase + e;
            int n = j * 16 + r;
            ctx[((size_t)(b * 2048 + m)) * 1024 + h * 64 + n] = (_Float16)acc[j][e];
        }
}

// ---------------------------------------------------------------------------
// Output GEMM: Out[4096,1024](f32) = ctx(f16) @ W_o + b_o, double-buffered.
__device__ __forceinline__ void outA_load(const _Float16* __restrict__ Ctx,
                                          int bm, int kk, int tid, v8h rc[2]) {
    #pragma unroll
    for (int t = 0; t < 2; t++) {
        int idx = tid + t * 256;                  // 512 x v8h
        int m = idx >> 2, c = (idx & 3) << 3;
        rc[t] = *(const v8h*)&Ctx[(size_t)(bm + m) * 1024 + kk + c];
    }
}
__device__ __forceinline__ void outA_stage(_Float16* As, int tid, const v8h rc[2]) {
    #pragma unroll
    for (int t = 0; t < 2; t++) {
        int idx = tid + t * 256;
        int m = idx >> 2, c = (idx & 3) << 3;
        *(v8h*)&As[m * 32 + c] = rc[t];           // 16B aligned
    }
}

__global__ __launch_bounds__(256) void out_kernel(
    const _Float16* __restrict__ Ctx, const float* __restrict__ W,
    const float* __restrict__ bias, float* __restrict__ Out)
{
    __shared__ alignas(16) _Float16 As[128 * 32];
    __shared__ alignas(16) _Float16 Bs[128 * 32];
    const int tid = threadIdx.x, lane = tid & 31, wave = tid >> 5;
    const int bm = blockIdx.y * 128, bn = blockIdx.x * 128;
    const int wm = (wave >> 1) * 32, wn = (wave & 1) * 64;
    const int r = lane & 15;
    const int ka = (lane >> 4) * 8, kbv = (lane >> 4) * 8;

    v8f acc[2][4];
    v8f vz = {};
    #pragma unroll
    for (int i = 0; i < 2; i++)
        #pragma unroll
        for (int j = 0; j < 4; j++) acc[i][j] = vz;

    v8h rc0[2], rc1[2];
    v4f rb0[4], rb1[4];
    outA_load(Ctx, bm, 0, tid, rc0);
    w_load(W, bn, 0, tid, rb0);

    for (int kk = 0; kk < 1024; kk += 64) {
        outA_stage(As, tid, rc0);
        w_stage(Bs, tid, rb0);
        __syncthreads();
        outA_load(Ctx, bm, kk + 32, tid, rc1);
        w_load(W, bn, kk + 32, tid, rb1);
        mma_2x4(As, Bs, wm, wn, r, ka, kbv, acc);
        __syncthreads();
        outA_stage(As, tid, rc1);
        w_stage(Bs, tid, rb1);
        __syncthreads();
        if (kk + 64 < 1024) {
            outA_load(Ctx, bm, kk + 64, tid, rc0);
            w_load(W, bn, kk + 64, tid, rb0);
        }
        mma_2x4(As, Bs, wm, wn, r, ka, kbv, acc);
        __syncthreads();
    }

    const int rbase = (lane >> 4) * 8;
    #pragma unroll
    for (int i = 0; i < 2; i++)
        #pragma unroll
        for (int j = 0; j < 4; j++)
            #pragma unroll
            for (int e = 0; e < 8; e++) {
                int m = bm + wm + i * 16 + rbase + e;
                int n = bn + wn + j * 16 + r;
                Out[(size_t)m * 1024 + n] = acc[i][j][e] + bias[n];
            }
}

// ---------------------------------------------------------------------------
extern "C" void kernel_launch(void* const* d_in, const int* in_sizes, int n_in,
                              void* d_out, int out_size, void* d_ws, size_t ws_size,
                              hipStream_t stream) {
    const float* q   = (const float*)d_in[0];
    const float* k   = (const float*)d_in[1];
    const float* v   = (const float*)d_in[2];
    const float* W_q = (const float*)d_in[3];
    const float* b_q = (const float*)d_in[4];
    const float* W_k = (const float*)d_in[5];
    const float* b_k = (const float*)d_in[6];
    const float* W_v = (const float*)d_in[7];
    const float* b_v = (const float*)d_in[8];
    const float* W_o = (const float*)d_in[9];
    const float* b_o = (const float*)d_in[10];

    float* out  = (float*)d_out;                      // [2,2048,1024]
    float* attn = out + (size_t)2 * 2048 * 1024;      // [2,16,2048,2048]

    const size_t headElems = (size_t)2 * 16 * 2048 * 64;
    _Float16* Qh  = (_Float16*)d_ws;                  // [B*H, S, 64] f16, Q/8
    _Float16* Kh  = Qh + headElems;
    _Float16* Vh  = Kh + headElems;
    _Float16* ctx = Vh + headElems;                   // [B, S, 1024] f16
    (void)ws_size; (void)in_sizes; (void)n_in; (void)out_size;

    dim3 blk(256);
    dim3 gProj(1024 / 128, 4096 / 128);
    proj_kernel<<<gProj, blk, 0, stream>>>(q, W_q, b_q, 0.125f, Qh);  // 1/sqrt(64)
    proj_kernel<<<gProj, blk, 0, stream>>>(k, W_k, b_k, 1.0f, Kh);
    proj_kernel<<<gProj, blk, 0, stream>>>(v, W_v, b_v, 1.0f, Vh);

    dim3 gSc(2048 / 128, 2048 / 128, 32);
    scores_kernel<<<gSc, blk, 0, stream>>>(Qh, Kh, attn);

    softmax_kernel<<<dim3(2 * 16 * 2048), blk, 0, stream>>>(attn);

    dim3 gAV(2048 / 128, 1, 32);
    av_kernel<<<gAV, blk, 0, stream>>>(attn, Vh, ctx);

    dim3 gOut(1024 / 128, 4096 / 128);
    out_kernel<<<gOut, blk, 0, stream>>>(ctx, W_o, b_o, out);
}
